// Model_14585708937301
// MI455X (gfx1250) — matrix operand order repaired and checked by reading, not compiled, not run on hardware
//
#include <hip/hip_runtime.h>
#include <hip/hip_bf16.h>

// ---------------------------------------------------------------------------
// CDNA5 (gfx1250) temporal-GraphSAGE. wave32, WMMA bf16 16x16x32, f32 accum.
// ---------------------------------------------------------------------------

typedef __attribute__((ext_vector_type(16))) __bf16 v16bf;
typedef __attribute__((ext_vector_type(8)))  float  v8f;

#define C_DIM 128
#define N_LAYERS 2
#define N_PERIODS 4

struct U16x16 { unsigned short u[16]; };

__device__ __forceinline__ unsigned short f2bf(float f) {
  // round-to-nearest-even f32 -> bf16
  unsigned u = __builtin_bit_cast(unsigned, f);
  unsigned r = u + 0x7fffu + ((u >> 16) & 1u);
  return (unsigned short)(r >> 16);
}

// Build a 16-element bf16 fragment from two 16-byte runs (LDS or global).
__device__ __forceinline__ v16bf frag_from(const unsigned short* s0,
                                           const unsigned short* s1) {
  U16x16 t;
  *reinterpret_cast<uint4*>(&t.u[0]) = *reinterpret_cast<const uint4*>(s0);
  *reinterpret_cast<uint4*>(&t.u[8]) = *reinterpret_cast<const uint4*>(s1);
  return __builtin_bit_cast(v16bf, t);
}

// ---------------------------------------------------------------- utilities
__global__ void zero_u32_kernel(unsigned* p, long n) {
  long i = (long)blockIdx.x * blockDim.x + threadIdx.x;
  long stride = (long)gridDim.x * blockDim.x;
  for (; i < n; i += stride) p[i] = 0u;
}

__global__ void copy_f4_kernel(float4* dst, const float4* src, long n4) {
  long i = (long)blockIdx.x * blockDim.x + threadIdx.x;
  long stride = (long)gridDim.x * blockDim.x;
  for (; i < n4; i += stride) dst[i] = src[i];
}

// Pack w_root/w_nbr into the per-lane B-fragment layout:
// idx = ((((l*2+mm)*4 + kc)*C + col)*2 + half)*16 + i  ;  k = kc*32 + half*16 + i
__global__ void pack_w_kernel(const float* __restrict__ wr,
                              const float* __restrict__ wn,
                              unsigned short* __restrict__ wpk) {
  int id = blockIdx.x * blockDim.x + threadIdx.x;
  const int total = N_LAYERS * 2 * 4 * C_DIM * 2 * 16;
  if (id >= total) return;
  int i = id & 15;      int t = id >> 4;
  int half = t & 1;     t >>= 1;
  int col = t % C_DIM;  t /= C_DIM;
  int kc = t & 3;       t >>= 2;
  int mm = t & 1;       t >>= 1;
  int l = t;
  int k = kc * 32 + half * 16 + i;
  const float* w = mm ? wn : wr;
  wpk[id] = f2bf(w[(size_t)l * C_DIM * C_DIM + (size_t)k * C_DIM + col]);
}

// Per-edge period + per-period degree / involvement counts.
__global__ void classify_kernel(const int* __restrict__ nt,
                                const int* __restrict__ ei,
                                int* __restrict__ eper,
                                int* __restrict__ degi,
                                int* __restrict__ cnt,
                                const int* __restrict__ min_ts,
                                const int* __restrict__ upd,
                                int n, int e) {
  int id = blockIdx.x * blockDim.x + threadIdx.x;
  if (id >= e) return;
  int s = ei[id], d = ei[e + id];
  int ts = nt[s], td = nt[d];
  int intro = ts > td ? ts : td;
  int u = upd[0]; if (u < 1) u = 1;
  int p = (intro - min_ts[0]) / u;
  if (p < 0) p = 0;
  if (p >= N_PERIODS) p = N_PERIODS - 1;
  eper[id] = p;
  atomicAdd(&degi[p * n + d], 1);
  atomicAdd(&cnt[p * n + s], 1);
  atomicAdd(&cnt[p * n + d], 1);
}

__global__ void dinv_kernel(const int* __restrict__ degi,
                            float* __restrict__ dinv, long n) {
  long i = (long)blockIdx.x * blockDim.x + threadIdx.x;
  if (i >= n) return;
  float d = (float)degi[i];
  dinv[i] = 1.0f / fmaxf(d, 1.0f);
}

// One wave per edge; each lane moves float4 of the 128-channel row.
__global__ void scatter_kernel(const float* __restrict__ hin,
                               float* __restrict__ agg,
                               const int* __restrict__ ei,
                               const int* __restrict__ eper,
                               int e, int p) {
  int gid = blockIdx.x * blockDim.x + threadIdx.x;
  int edge = gid >> 5;
  int lane = gid & 31;
  if (edge >= e) return;
  if (eper[edge] != p) return;
  int s = ei[edge], d = ei[e + edge];
  const float4 v =
      *reinterpret_cast<const float4*>(hin + (size_t)s * C_DIM + lane * 4);
  float* ad = agg + (size_t)d * C_DIM + lane * 4;
  __hip_atomic_fetch_add(ad + 0, v.x, __ATOMIC_RELAXED, __HIP_MEMORY_SCOPE_AGENT);
  __hip_atomic_fetch_add(ad + 1, v.y, __ATOMIC_RELAXED, __HIP_MEMORY_SCOPE_AGENT);
  __hip_atomic_fetch_add(ad + 2, v.z, __ATOMIC_RELAXED, __HIP_MEMORY_SCOPE_AGENT);
  __hip_atomic_fetch_add(ad + 3, v.w, __ATOMIC_RELAXED, __HIP_MEMORY_SCOPE_AGENT);
}

// Fused: [h | agg*invdeg] x [w_root | w_nbr] + bias -> LayerNorm (-> ReLU or
// masked residual into xbuf). 16-node tile per block, 8 waves, one 16-col WMMA
// tile per wave, K=128 as 4 chunks of 32 (bf16 inputs, f32 accumulation).
__global__ __launch_bounds__(256) void gemm_ln_kernel(
    const float* __restrict__ hin, const float* __restrict__ agg,
    const float* __restrict__ dinv_p, const unsigned short* __restrict__ wpk_l,
    const float* __restrict__ bias_l, const float* __restrict__ g_l,
    const float* __restrict__ b_l, float* __restrict__ hout,
    float* __restrict__ xbuf, const int* __restrict__ cnt_p,
    int n, int mode /*0: h1 = relu(ln(..)); 1: xbuf += ln(..) if involved*/) {
  __shared__ unsigned short sa_r[16 * C_DIM];   // root-input tile, bf16
  __shared__ unsigned short sa_n[16 * C_DIM];   // scaled-aggregate tile, bf16
  __shared__ float sout[16 * C_DIM];            // f32 accum tile
  __shared__ float red_s[256];
  __shared__ float red_q[256];
  __shared__ float smu[16];
  __shared__ float srs[16];

  const int tid = threadIdx.x;
  const int tile0 = blockIdx.x * 16;

  // ---- stage A tiles into LDS as bf16 ----
  {
    const int r = tid >> 4;
    const int cb = (tid & 15) * 8;
    int node = tile0 + r;
    if (node >= n) node = n - 1;
    const size_t base = (size_t)node * C_DIM + cb;
    const float4 h0 = *reinterpret_cast<const float4*>(hin + base);
    const float4 h1v = *reinterpret_cast<const float4*>(hin + base + 4);
    const float idg = dinv_p[node];
    const float4 a0 = *reinterpret_cast<const float4*>(agg + base);
    const float4 a1 = *reinterpret_cast<const float4*>(agg + base + 4);
    unsigned short* pr = &sa_r[r * C_DIM + cb];
    unsigned short* pn = &sa_n[r * C_DIM + cb];
    pr[0] = f2bf(h0.x);  pr[1] = f2bf(h0.y);
    pr[2] = f2bf(h0.z);  pr[3] = f2bf(h0.w);
    pr[4] = f2bf(h1v.x); pr[5] = f2bf(h1v.y);
    pr[6] = f2bf(h1v.z); pr[7] = f2bf(h1v.w);
    pn[0] = f2bf(a0.x * idg); pn[1] = f2bf(a0.y * idg);
    pn[2] = f2bf(a0.z * idg); pn[3] = f2bf(a0.w * idg);
    pn[4] = f2bf(a1.x * idg); pn[5] = f2bf(a1.y * idg);
    pn[6] = f2bf(a1.z * idg); pn[7] = f2bf(a1.w * idg);
  }
  __syncthreads();

  const int lane = tid & 31;
  const int wv = tid >> 5;
  const int halfId = lane >> 4;
  const int m = lane & 15;
  const int col = wv * 16 + m;

  v8f acc = {};
#pragma unroll
  for (int kc = 0; kc < 4; ++kc) {
    // A fragments: lane row m; contiguous 16-K half per half-wave (k = 16h + s).
    v16bf ar = frag_from(&sa_r[m * C_DIM + kc * 32 + halfId * 16],
                         &sa_r[m * C_DIM + kc * 32 + halfId * 16 + 8]);
    v16bf an = frag_from(&sa_n[m * C_DIM + kc * 32 + halfId * 16],
                         &sa_n[m * C_DIM + kc * 32 + halfId * 16 + 8]);
    // B fragments: pre-packed contiguous 32B per lane.
    const unsigned short* wr =
        wpk_l + ((((size_t)(0 * 4 + kc) * C_DIM + col) * 2 + halfId) * 16);
    const unsigned short* wn =
        wpk_l + ((((size_t)(1 * 4 + kc) * C_DIM + col) * 2 + halfId) * 16);
    v16bf br = frag_from(wr, wr + 8);
    v16bf bn = frag_from(wn, wn + 8);
    acc = __builtin_amdgcn_wmma_f32_16x16x32_bf16(false, ar, false, br,
                                                  (short)0, acc, false, false);
    acc = __builtin_amdgcn_wmma_f32_16x16x32_bf16(false, an, false, bn,
                                                  (short)0, acc, false, false);
  }

  // acc[j] holds row (j + 8*halfId), column col. Add bias, spill to LDS.
  const float bc = bias_l[col];
#pragma unroll
  for (int j = 0; j < 8; ++j)
    sout[(j + 8 * halfId) * C_DIM + col] = acc[j] + bc;
  __syncthreads();

  // ---- LayerNorm over the 128 channels of each row ----
  {
    const int r = tid >> 4;
    const int cb = (tid & 15) * 8;
    float s = 0.f, q = 0.f;
#pragma unroll
    for (int j = 0; j < 8; ++j) {
      float v = sout[r * C_DIM + cb + j];
      s += v;
      q += v * v;
    }
    red_s[tid] = s;
    red_q[tid] = q;
  }
  __syncthreads();
  if ((tid & 15) == 0) {
    const int r = tid >> 4;
    float s = 0.f, q = 0.f;
#pragma unroll
    for (int j = 0; j < 16; ++j) {
      s += red_s[(r << 4) + j];
      q += red_q[(r << 4) + j];
    }
    const float mu = s * (1.0f / C_DIM);
    const float var = q * (1.0f / C_DIM) - mu * mu;
    smu[r] = mu;
    srs[r] = rsqrtf(var + 1e-5f);
  }
  __syncthreads();
  {
    const int r = tid >> 4;
    const int cb = (tid & 15) * 8;
    const int node = tile0 + r;
    if (node < n) {
      const float mu = smu[r], rs = srs[r];
      if (mode == 0) {
        float* dst = hout + (size_t)node * C_DIM + cb;
#pragma unroll
        for (int j = 0; j < 8; ++j) {
          float v = (sout[r * C_DIM + cb + j] - mu) * rs * g_l[cb + j] + b_l[cb + j];
          dst[j] = fmaxf(v, 0.f);
        }
      } else if (cnt_p[node] > 0) {
        float* dst = xbuf + (size_t)node * C_DIM + cb;
#pragma unroll
        for (int j = 0; j < 8; ++j) {
          float v = (sout[r * C_DIM + cb + j] - mu) * rs * g_l[cb + j] + b_l[cb + j];
          dst[j] += v;
        }
      }
    }
  }
}

// Head: one wave per seed row, OUT == 1.
__global__ void head_kernel(const float* __restrict__ xbuf,
                            const float* __restrict__ hw,
                            const float* __restrict__ hb,
                            float* __restrict__ out, int seeds) {
  int gid = blockIdx.x * blockDim.x + threadIdx.x;
  int s = gid >> 5;
  int lane = gid & 31;
  if (s >= seeds) return;
  const float4 a =
      *reinterpret_cast<const float4*>(xbuf + (size_t)s * C_DIM + lane * 4);
  const float4 b = *reinterpret_cast<const float4*>(hw + lane * 4);
  float d = a.x * b.x + a.y * b.y + a.z * b.z + a.w * b.w;
#pragma unroll
  for (int off = 16; off > 0; off >>= 1) d += __shfl_xor(d, off, 32);
  if (lane == 0) out[s] = d + hb[0];
}

// ---------------------------------------------------------------------------
extern "C" void kernel_launch(void* const* d_in, const int* in_sizes, int n_in,
                              void* d_out, int out_size, void* d_ws,
                              size_t ws_size, hipStream_t stream) {
  const float* x         = (const float*)d_in[0];
  const int*   node_time = (const int*)d_in[1];
  const int*   edge_idx  = (const int*)d_in[2];
  const float* w_root    = (const float*)d_in[3];
  const float* w_nbr     = (const float*)d_in[4];
  const float* bias      = (const float*)d_in[5];
  const float* ln_g      = (const float*)d_in[6];
  const float* ln_b      = (const float*)d_in[7];
  const float* head_w    = (const float*)d_in[8];
  const float* head_b    = (const float*)d_in[9];
  const int*   min_ts    = (const int*)d_in[10];
  const int*   upd       = (const int*)d_in[12];

  const int N = in_sizes[1];
  const int E = in_sizes[2] / 2;
  const int P = N_PERIODS;          // (3 - 0 + 1)/1 from reference constants
  const int seeds = out_size;       // OUT == 1

  // ---- workspace carve-out ----
  uintptr_t base = (uintptr_t)d_ws;
  auto carve = [&](size_t bytes) -> void* {
    void* p = (void*)base;
    base += (bytes + 255) & ~(size_t)255;
    return p;
  };
  float* xbuf = (float*)carve((size_t)N * C_DIM * 4);
  float* h1   = (float*)carve((size_t)N * C_DIM * 4);
  float* agg  = (float*)carve((size_t)N * C_DIM * 4);
  int*   eper = (int*)carve((size_t)E * 4);
  int*   degi = (int*)carve((size_t)P * N * 4);
  int*   cnt  = (int*)carve((size_t)P * N * 4);
  float* dinv = (float*)carve((size_t)P * N * 4);
  unsigned short* wpk =
      (unsigned short*)carve((size_t)N_LAYERS * 2 * 4 * C_DIM * 2 * 16 * 2);

  const int ZB = 2048, TB = 256;

  // One-time (per call) setup.
  zero_u32_kernel<<<ZB, TB, 0, stream>>>((unsigned*)degi, (long)P * N);
  zero_u32_kernel<<<ZB, TB, 0, stream>>>((unsigned*)cnt, (long)P * N);
  copy_f4_kernel<<<ZB, TB, 0, stream>>>((float4*)xbuf, (const float4*)x,
                                        (long)N * C_DIM / 4);
  {
    const int total = N_LAYERS * 2 * 4 * C_DIM * 2 * 16;
    pack_w_kernel<<<(total + TB - 1) / TB, TB, 0, stream>>>(w_root, w_nbr, wpk);
  }
  classify_kernel<<<(E + TB - 1) / TB, TB, 0, stream>>>(
      node_time, edge_idx, eper, degi, cnt, min_ts, upd, N, E);
  dinv_kernel<<<((long)P * N + TB - 1) / TB, TB, 0, stream>>>(degi, dinv,
                                                              (long)P * N);

  const int scat_blocks = (int)(((long)E * 32 + TB - 1) / TB);
  const int gemm_blocks = (N + 15) / 16;

  for (int p = 0; p < P; ++p) {
    const float* dinv_p = dinv + (size_t)p * N;
    const int* cnt_p = cnt + (size_t)p * N;

    // Layer 0: input = xbuf
    zero_u32_kernel<<<ZB, TB, 0, stream>>>((unsigned*)agg, (long)N * C_DIM);
    scatter_kernel<<<scat_blocks, TB, 0, stream>>>(xbuf, agg, edge_idx, eper,
                                                   E, p);
    gemm_ln_kernel<<<gemm_blocks, TB, 0, stream>>>(
        xbuf, agg, dinv_p, wpk + (size_t)0 * 2 * 4 * C_DIM * 2 * 16,
        bias + 0 * C_DIM, ln_g + 0 * C_DIM, ln_b + 0 * C_DIM, h1, xbuf, cnt_p,
        N, /*mode=*/0);

    // Layer 1: input = h1, fused residual update of xbuf (involved only)
    zero_u32_kernel<<<ZB, TB, 0, stream>>>((unsigned*)agg, (long)N * C_DIM);
    scatter_kernel<<<scat_blocks, TB, 0, stream>>>(h1, agg, edge_idx, eper, E,
                                                   p);
    gemm_ln_kernel<<<gemm_blocks, TB, 0, stream>>>(
        h1, agg, dinv_p, wpk + (size_t)1 * 2 * 4 * C_DIM * 2 * 16,
        bias + 1 * C_DIM, ln_g + 1 * C_DIM, ln_b + 1 * C_DIM, h1, xbuf, cnt_p,
        N, /*mode=*/1);
  }

  head_kernel<<<(seeds * 32 + TB - 1) / TB, TB, 0, stream>>>(
      xbuf, head_w, head_b, (float*)d_out, seeds);
}
